// VectorQuantizer2_61632780697604
// MI455X (gfx1250) — matrix operand order repaired and checked by reading, not compiled, not run on hardware
//
#include <hip/hip_runtime.h>
#include <hip/hip_bf16.h>
#include <cstdint>

// ---------------------------------------------------------------------------
// Problem constants
// ---------------------------------------------------------------------------
#define D_DIM   256          // latent dim (= GEMM K)
#define K_CODES 8192         // codebook size
#define HW      1024         // H*W
#define N_PIX   32768        // B*H*W
#define MBLK    64           // rows of zf per block
#define ROWP    264          // padded row length (bf16) -> 528B row, conflict-free b128

typedef __attribute__((ext_vector_type(16))) __bf16 v16bf;
typedef __attribute__((ext_vector_type(8)))  float  v8f;

// TDM descriptor vector types (probe-verified builtin signature)
typedef __attribute__((ext_vector_type(4))) unsigned int tdm_u32x4;
typedef __attribute__((ext_vector_type(8))) int          tdm_i32x8;
typedef __attribute__((ext_vector_type(4))) int          tdm_i32x4;

#if defined(__AMDGCN__) && __has_builtin(__builtin_amdgcn_tensor_load_to_lds)
#define HAVE_TDM 1
#else
#define HAVE_TDM 0
#endif

#if HAVE_TDM
// DMA one 32-row x 256-elem bf16 tile pair (16 KB contiguous in global) into
// LDS with 16B padding appended to every 512B row (ROWP=264 layout).
__device__ __forceinline__ void tdm_load_pair(const __bf16* gsrc, void* ldst)
{
    unsigned long long ga = (unsigned long long)(uintptr_t)gsrc;
    unsigned int       la = (unsigned int)(uintptr_t)ldst;   // low 32 bits = LDS offset

    tdm_u32x4 g0;
    g0.x = 1u;                                   // count=1, user D#, no gather
    g0.y = la;                                   // lds_addr
    g0.z = (unsigned int)ga;                     // global_addr[31:0]
    g0.w = (unsigned int)(ga >> 32) | (2u << 30);// global_addr[56:32] | type=2 (image)

    tdm_i32x8 g1;
    g1[0] = (1 << 16)        // data_size = 2 bytes
          | (1 << 20)        // pad_enable
          | (6 << 22)        // pad_interval: every 128 DWORDs (=512B row)
          | (3 << 25);       // pad_amount: 4 DWORDs (=16B)
    g1[1] = (int)(256u << 16);       // tensor_dim0 = 256 elems (bits 79:48 lo)
    g1[2] = (int)(8192u << 16);      // tensor_dim1 = 8192 rows (bits 111:80 lo)
    g1[3] = (int)(256u << 16);       // tile_dim0 = 256 elems (bits 127:112)
    g1[4] = 32;                      // tile_dim1 = 32 rows; tile_dim2 = 0
    g1[5] = 256;                     // tensor_dim0_stride = 256 elems
    g1[6] = (int)(256u << 16);       // tensor_dim1_stride lo (unused for 2D tile)
    g1[7] = 0;

    tdm_i32x4 z4 = {0, 0, 0, 0};
#if __clang_major__ >= 23
    tdm_i32x8 z8 = {0, 0, 0, 0, 0, 0, 0, 0};
    __builtin_amdgcn_tensor_load_to_lds(g0, g1, z4, z4, z8, 0);
#else
    __builtin_amdgcn_tensor_load_to_lds(g0, g1, z4, z4, 0);
#endif
}

__device__ __forceinline__ void tdm_wait0()
{
#if __has_builtin(__builtin_amdgcn_s_wait_tensorcnt)
    __builtin_amdgcn_s_wait_tensorcnt(0);
#else
    asm volatile("s_wait_tensorcnt 0x0" ::: "memory");
#endif
}
#endif // HAVE_TDM

// ---------------------------------------------------------------------------
// Kernel 1: emb (K,D) fp32 -> bf16 copy + per-code squared norms
// ---------------------------------------------------------------------------
__global__ __launch_bounds__(256)
void vq_prep_emb(const float* __restrict__ emb,
                 __bf16* __restrict__ embB,
                 float* __restrict__ enorm)
{
    const int k = blockIdx.x;
    const int d = threadIdx.x;
    float v = emb[(size_t)k * D_DIM + d];
    embB[(size_t)k * D_DIM + d] = (__bf16)v;

    __shared__ float red[256];
    red[d] = v * v;
    __syncthreads();
#pragma unroll
    for (int s = 128; s > 0; s >>= 1) {
        if (d < s) red[d] += red[d + s];
        __syncthreads();
    }
    if (d == 0) enorm[k] = red[0];
}

// ---------------------------------------------------------------------------
// Kernel 2: fused distance GEMM + argmin.
//   d_rel[n,c] = ||e_c||^2 + (-2 z_n) . e_c
//   Phase-aliased 33 KB LDS: A slab -> 4 double-buffered B tiles -> argmin scratch
//   Wave w: m-subtile (w&3), code-parity group (w>>2). TDM streams B tiles.
// ---------------------------------------------------------------------------
__global__ __launch_bounds__(256)
void vq_argmin(const float* __restrict__ z,
               const __bf16* __restrict__ embB,
               const float* __restrict__ enorm,
               float* __restrict__ idxOutF,
               int*   __restrict__ idxOutI)
{
    __shared__ __align__(16) char smem[4 * 16 * ROWP * 2];      // 33792 B
    __bf16 (*shA)[ROWP]     = (__bf16 (*)[ROWP])smem;           // phase 1: 64x264 A slab
    __bf16 (*shB)[16][ROWP] = (__bf16 (*)[16][ROWP])smem;       // phase 2: 4 B tiles
    float* redV = (float*)smem;                                 // phase 3: merge scratch
    int*   redI = (int*)(smem + MBLK * 32 * sizeof(float));

    const int tid  = threadIdx.x;
    const int wave = tid >> 5;
    const int lane = tid & 31;

    const int nBase = blockIdx.x * MBLK;     // 64 consecutive pixels, same image b
    const int bImg  = nBase >> 10;
    const int hw0   = nBase & (HW - 1);

    // ---- phase 1: stage A slab shA[r][d] = bf16(-2 * z[b, d, hw0+r]) ----
    const float* zBase = z + (size_t)bImg * D_DIM * HW + hw0;
    for (int idx = tid; idx < MBLK * D_DIM; idx += 256) {
        int r = idx & (MBLK - 1);            // coalesced along pixels
        int d = idx >> 6;
        float v = zBase[(size_t)d * HW + r];
        shA[r][d] = (__bf16)(-2.0f * v);
    }
    __syncthreads();

    const int g    = wave >> 2;              // code-parity group (0/1)
    const int m0   = (wave & 3) << 4;        // m-subtile base row
    const int hi   = lane >> 4;
    const int rB   = lane & 15;
    const int rowA = m0 + rB;

    union F16x16 { v16bf v; uint4 q[2]; };

    // ---- hoist A fragments into registers (ISA 16-bit A 16x32 lane map) ----
    v16bf afrag[8];
#pragma unroll
    for (int kc = 0; kc < 8; ++kc) {
        F16x16 u;
        u.q[0] = *(const uint4*)&shA[rowA][kc * 32 + hi * 8];
        u.q[1] = *(const uint4*)&shA[rowA][kc * 32 + 16 + hi * 8];
        afrag[kc] = u.v;
    }
    __syncthreads();                         // A slab dead -> reuse as B buffers

    float bmin[8];
    int   bidx[8];
#pragma unroll
    for (int r = 0; r < 8; ++r) { bmin[r] = 3.4e38f; bidx[r] = 0; }

    const int NT = K_CODES / 32;             // 256 tile-pair iterations

    // ---- prologue: bring in pair 0 (buffers 0,1) ----
#if HAVE_TDM
    if (wave == 0) {
        tdm_load_pair(embB, smem);
        tdm_wait0();
    }
#else
    for (int e = tid; e < 1024; e += 256) {
        int row = e >> 5, off = e & 31;
        uint4 val = *(const uint4*)(embB + (size_t)row * D_DIM + off * 8);
        *(uint4*)&shB[row >> 4][row & 15][off * 8] = val;
    }
#endif
    __syncthreads();

    for (int j = 0; j < NT; ++j) {
        const int cb = (j & 1) << 1;         // current pair buffer base

        // ---- prefetch pair j+1 into the idle buffers (overlaps compute) ----
#if HAVE_TDM
        if (j + 1 < NT && wave == 0)
            tdm_load_pair(embB + (size_t)(j + 1) * 32 * D_DIM,
                          smem + (size_t)(((j + 1) & 1) << 1) * 16 * ROWP * 2);
#else
        if (j + 1 < NT) {
            const int pb = ((j + 1) & 1) << 1;
            for (int e = tid; e < 1024; e += 256) {
                int row = e >> 5, off = e & 31;
                uint4 val = *(const uint4*)(embB +
                    (size_t)((j + 1) * 32 + row) * D_DIM + off * 8);
                *(uint4*)&shB[pb + (row >> 4)][row & 15][off * 8] = val;
            }
            if (j + 2 < NT)
                __builtin_prefetch(embB + (size_t)(j + 2) * 32 * D_DIM + tid * 32, 0, 1);
        }
#endif

        // ---- batch-load 8 B fragments (ISA 16-bit B 32x16 lane map) ----
        const __bf16 (*bt)[ROWP] = shB[cb + g];
        v16bf bfrag[8];
#pragma unroll
        for (int kc = 0; kc < 8; ++kc) {
            F16x16 ub;
            ub.q[0] = *(const uint4*)&bt[rB][kc * 32 + hi * 8];
            ub.q[1] = *(const uint4*)&bt[rB][kc * 32 + 16 + hi * 8];
            bfrag[kc] = ub.v;
        }

        v8f acc = {0.f, 0.f, 0.f, 0.f, 0.f, 0.f, 0.f, 0.f};
#pragma unroll
        for (int kc = 0; kc < 8; ++kc)
            acc = __builtin_amdgcn_wmma_f32_16x16x32_bf16(
                false, afrag[kc], false, bfrag[kc], (short)0, acc, false, false);

        const int   code = j * 32 + g * 16 + rB;   // C/D col N = lane&15
        const float en   = enorm[code];
#pragma unroll
        for (int r = 0; r < 8; ++r) {
            float v = acc[r] + en;
            if (v < bmin[r]) { bmin[r] = v; bidx[r] = code; }
        }

#if HAVE_TDM
        if (wave == 0) tdm_wait0();          // pair j+1 landed
#endif
        __syncthreads();                     // all waves done with pair j
    }

    // ---- merge argmin across 16 lanes x 2 parity groups per row ----
#pragma unroll
    for (int r = 0; r < 8; ++r) {
        int row  = m0 + r + 8 * hi;          // C row M = r + 8*(lane>>4)
        int slot = g * 16 + rB;
        redV[row * 32 + slot] = bmin[r];
        redI[row * 32 + slot] = bidx[r];
    }
    __syncthreads();

    if (tid < MBLK) {
        float best = 3.4e38f;
        int   bi   = K_CODES;
#pragma unroll 4
        for (int s = 0; s < 32; ++s) {
            float v = redV[tid * 32 + s];
            int   i = redI[tid * 32 + s];
            if (v < best || (v == best && i < bi)) { best = v; bi = i; }
        }
        int n = nBase + tid;
        idxOutF[n] = (float)bi;              // indices output (as float)
        idxOutI[n] = bi;                     // int copy for gather kernel
    }
}

// ---------------------------------------------------------------------------
// Kernel 3: gather z_q into (B,C,H,W) + fused 1.25 * MSE(z_q, z) loss
// ---------------------------------------------------------------------------
__global__ __launch_bounds__(256)
void vq_output(const float* __restrict__ z,
               const float* __restrict__ emb,
               const int*   __restrict__ idxI,
               float* __restrict__ qout,
               float* __restrict__ loss)
{
    const int bh = blockIdx.x;               // b*32 + h
    const int b  = bh >> 5;
    const int h  = bh & 31;
    const int t  = threadIdx.x;
    const int w  = t & 31;
    const int cg = t >> 5;                   // uniform per wave -> coalesced rows

    const int n    = (bh << 5) + w;
    const int code = idxI[n];
    const float* erow = emb + (size_t)code * D_DIM;

    float part = 0.f;
#pragma unroll 4
    for (int c = cg; c < D_DIM; c += 8) {
        size_t off = ((size_t)(b * D_DIM + c)) * HW + (h << 5) + w;
        float e  = erow[c];
        float zv = z[off];
        qout[off] = e;
        float d = e - zv;
        part += d * d;
    }

    __shared__ float red[256];
    red[t] = part;
    __syncthreads();
#pragma unroll
    for (int s = 128; s > 0; s >>= 1) {
        if (t < s) red[t] += red[t + s];
        __syncthreads();
    }
    // loss = (1 + COMMIT_W) * mean = 1.25 * sum / (N_PIX * D_DIM)
    if (t == 0) atomicAdd(loss, red[0] * (1.25f / 8388608.0f));
}

// ---------------------------------------------------------------------------
// Launcher
// ---------------------------------------------------------------------------
extern "C" void kernel_launch(void* const* d_in, const int* in_sizes, int n_in,
                              void* d_out, int out_size, void* d_ws, size_t ws_size,
                              hipStream_t stream)
{
    const float* z   = (const float*)d_in[0];   // (32,256,32,32) fp32
    const float* emb = (const float*)d_in[1];   // (8192,256) fp32

    float* out = (float*)d_out;
    const size_t QELEMS = (size_t)32 * 256 * 32 * 32;   // 8388608
    float* idxOutF = out + QELEMS;                      // 32768 indices (as float)
    float* lossOut = out + QELEMS + N_PIX;              // scalar loss

    // workspace: embB (4 MB) | enorm (32 KB) | idxI (128 KB)
    char* ws = (char*)d_ws;
    __bf16* embB = (__bf16*)ws;
    float*  enorm = (float*)(ws + (size_t)K_CODES * D_DIM * sizeof(__bf16));
    int*    idxI  = (int*)(ws + (size_t)K_CODES * D_DIM * sizeof(__bf16)
                              + (size_t)K_CODES * sizeof(float));

    vq_prep_emb<<<K_CODES, 256, 0, stream>>>(emb, embB, enorm);
    vq_argmin<<<N_PIX / MBLK, 256, 0, stream>>>(z, embB, enorm, idxOutF, idxI);
    hipMemsetAsync(lossOut, 0, sizeof(float), stream);
    vq_output<<<32 * 32, 256, 0, stream>>>(z, emb, idxI, out, lossOut);
}
